// NanoporeRVQModel_91104846282892
// MI455X (gfx1250) — matrix operand order repaired and checked by reading, not compiled, not run on hardware
//
#include <hip/hip_runtime.h>
#include <stdint.h>

typedef __attribute__((ext_vector_type(16))) __bf16 v16bf;
typedef __attribute__((ext_vector_type(8)))  float  v8f;
typedef __attribute__((ext_vector_type(4)))  unsigned int u32x4;
typedef __attribute__((ext_vector_type(8)))  int i32x8;
typedef __attribute__((ext_vector_type(4)))  int i32x4;
typedef unsigned short u16;
typedef unsigned int   u32;

#if defined(__has_builtin)
#if __has_builtin(__builtin_amdgcn_tensor_load_to_lds) && __has_builtin(__builtin_amdgcn_s_wait_tensorcnt)
#define HAVE_TDM 1
#endif
#endif
#ifndef HAVE_TDM
#define HAVE_TDM 0
#endif

union V16 { v16bf v; uint4 u[2]; };

__device__ __forceinline__ u16 f32_to_bf16(float f) {
  u32 u = __float_as_uint(f);
  u32 r = u + 0x7fffu + ((u >> 16) & 1u);
  return (u16)(r >> 16);
}
__device__ __forceinline__ float bf16_to_f32(u16 h) {
  return __uint_as_float(((u32)h) << 16);
}

// ---------------------------------------------------------------------------
// enc block 1: Cin=1 direct conv + bias + silu  (y fp32 token-major [tok][64])
// ---------------------------------------------------------------------------
__global__ void enc1_kernel(const float* __restrict__ X, const float* __restrict__ W,
                            const float* __restrict__ B, float* __restrict__ Y,
                            int L, int N, int Cout, int K, int pad)
{
  size_t total = (size_t)N * L * Cout;
  size_t i = (size_t)blockIdx.x * blockDim.x + threadIdx.x;
  if (i >= total) return;
  int co = (int)(i % Cout);
  size_t tok = i / Cout;
  int l = (int)(tok % L);
  int n = (int)(tok / L);
  float acc = B[co];
  for (int t = 0; t < K; ++t) {
    int li = l + t - pad;
    if (li >= 0 && li < L) acc += W[co * K + t] * X[(size_t)n * L + li];
  }
  acc = acc / (1.0f + __expf(-acc));
  Y[i] = acc;
}

// ---------------------------------------------------------------------------
// Weight repack: fp32 (O,I,K) or (I,O,K) -> bf16 [t][co][ci]
// ---------------------------------------------------------------------------
__global__ void wprep_kernel(const float* __restrict__ W, u16* __restrict__ Wb,
                             int Cout, int Cin, int K, int isT, size_t total)
{
  size_t i = (size_t)blockIdx.x * blockDim.x + threadIdx.x;
  if (i >= total) return;
  int ci = (int)(i % Cin);
  size_t r = i / Cin;
  int co = (int)(r % Cout);
  int t  = (int)(r / Cout);
  float v = isT ? W[((size_t)ci * Cout + co) * K + t]
                : W[((size_t)co * Cin + ci) * K + t];
  Wb[i] = f32_to_bf16(v);
}

// ---------------------------------------------------------------------------
// Implicit-GEMM conv / convT with WMMA bf16 -> f32.  D tile = 16 co x 16 tok,
// 4 accumulators per wave (16 co x 64 tok).  Epilogue: +bias, silu, fp32 store.
// Activations token-major bf16: X[tok][Cin].  Weights: Wb[t][co][ci].
// Layouts per CDNA5 ISA 7.12.2:
//   A (16x32, MxK): lane<16 -> K kb+0..7 & kb+16..23 of row M=lane;
//                   lane>=16 -> K kb+8..15 & kb+24..31 of row M=lane-16
//   B (32x16, KxN): lane<16 -> K kb+0..7 & kb+16..23 of col N=lane;
//                   lane>=16 -> K kb+8..15 & kb+24..31 of col N=lane-16
//   C/D (16x16 f32): lane<16 -> rows 0..7 of col lane; lane>=16 rows 8..15
// ---------------------------------------------------------------------------
__global__ void conv_wmma_kernel(const u16* __restrict__ X, const u16* __restrict__ Wb,
                                 const float* __restrict__ bias, float* __restrict__ Y,
                                 int Cin, int Cout, int Lin, int Lout, int totTok,
                                 int ktaps, int stride, int pad, int isT)
{
  int lane = threadIdx.x & 31;
  int wid  = (int)((blockIdx.x * blockDim.x + threadIdx.x) >> 5);
  int coTiles  = Cout >> 4;
  int tokTiles = (totTok + 63) >> 6;
  if (wid >= coTiles * tokTiles) return;
  int cot = wid % coTiles;
  int tkt = wid / coTiles;
  int co_base  = cot << 4;
  int tok_base = tkt << 6;
  int hi   = (lane >> 4) & 1;
  int lcol = lane & 15;

  v8f acc[4];
#pragma unroll
  for (int j = 0; j < 4; ++j) acc[j] = (v8f){0.f,0.f,0.f,0.f,0.f,0.f,0.f,0.f};

  int tokv[4], nn[4], lo[4];
  bool tv[4];
#pragma unroll
  for (int j = 0; j < 4; ++j) {
    tokv[j] = tok_base + j * 16 + lcol;
    tv[j] = tokv[j] < totTok;
    int tt = tv[j] ? tokv[j] : 0;
    nn[j] = tt / Lout;
    lo[j] = tt % Lout;
  }

  const int arow = co_base + lcol;
  for (int t = 0; t < ktaps; ++t) {
    const u16* Wt = Wb + (size_t)t * Cout * Cin + (size_t)arow * Cin;
    __builtin_prefetch(Wt + (size_t)Cout * Cin, 0, 1);   // next tap's weights
    const u16* xrow[4];
    bool bv[4];
#pragma unroll
    for (int j = 0; j < 4; ++j) {
      bool v = tv[j];
      int li = 0;
      if (isT) {
        int num = lo[j] + pad - t;
        v = v && (num >= 0) && (num % stride == 0);
        li = (num >= 0) ? (num / stride) : 0;
        v = v && (li < Lin);
      } else {
        li = lo[j] * stride + t - pad;
        v = v && (li >= 0) && (li < Lin);
      }
      bv[j] = v;
      xrow[j] = X + (size_t)(v ? (nn[j] * Lin + li) : 0) * Cin;
    }
    for (int kb = 0; kb < Cin; kb += 32) {
      V16 a;
      const u16* ab = Wt + kb + hi * 8;
      a.u[0] = *(const uint4*)(ab);
      a.u[1] = *(const uint4*)(ab + 16);
#pragma unroll
      for (int j = 0; j < 4; ++j) {
        V16 b;
        if (bv[j]) {
          const u16* bb = xrow[j] + kb + hi * 8;
          b.u[0] = *(const uint4*)(bb);
          b.u[1] = *(const uint4*)(bb + 16);
        } else {
          b.u[0] = make_uint4(0, 0, 0, 0);
          b.u[1] = make_uint4(0, 0, 0, 0);
        }
        acc[j] = __builtin_amdgcn_wmma_f32_16x16x32_bf16(
            false, a.v, false, b.v, (short)0, acc[j], false, false);
      }
    }
  }

  float bsv[8];
#pragma unroll
  for (int r = 0; r < 8; ++r) bsv[r] = bias[co_base + hi * 8 + r];
#pragma unroll
  for (int j = 0; j < 4; ++j) {
    if (!tv[j]) continue;
    float* yrow = Y + (size_t)tokv[j] * Cout + co_base + hi * 8;
#pragma unroll
    for (int r = 0; r < 8; ++r) {
      float v = acc[j][r] + bsv[r];
      v = v / (1.0f + __expf(-v));      // silu
      yrow[r] = v;
    }
  }
}

// ---------------------------------------------------------------------------
// BatchNorm over all tokens (N,L) per channel: stats then apply.
// ---------------------------------------------------------------------------
__global__ void bn_stats_kernel(const float* __restrict__ Y, float* __restrict__ stats,
                                int C, int T)
{
  __shared__ float s1[256], s2[256];
  int c = blockIdx.x;
  float a = 0.f, b = 0.f;
  for (int t = threadIdx.x; t < T; t += blockDim.x) {
    float v = Y[(size_t)t * C + c];
    a += v; b += v * v;
  }
  s1[threadIdx.x] = a; s2[threadIdx.x] = b;
  __syncthreads();
  for (int s = 128; s > 0; s >>= 1) {
    if ((int)threadIdx.x < s) {
      s1[threadIdx.x] += s1[threadIdx.x + s];
      s2[threadIdx.x] += s2[threadIdx.x + s];
    }
    __syncthreads();
  }
  if (threadIdx.x == 0) {
    float m   = s1[0] / (float)T;
    float var = s2[0] / (float)T - m * m;
    stats[c * 2]     = m;
    stats[c * 2 + 1] = rsqrtf(var + 1e-5f);
  }
}

__global__ void bn_apply_kernel(const float* __restrict__ Y, const float* __restrict__ stats,
                                const float* __restrict__ g, const float* __restrict__ beta,
                                u16* __restrict__ Ob, float* __restrict__ Of,
                                int C, size_t total)
{
  size_t i = (size_t)blockIdx.x * blockDim.x + threadIdx.x;
  if (i >= total) return;
  int c = (int)(i % C);
  float v = (Y[i] - stats[c * 2]) * stats[c * 2 + 1] * g[c] + beta[c];
  Ob[i] = f32_to_bf16(v);
  if (Of) Of[i] = v;
}

// ---------------------------------------------------------------------------
// RVQ: codebook prep (bf16 cast + squared norms)
// ---------------------------------------------------------------------------
__global__ void cbprep_kernel(const float* __restrict__ CB, u16* __restrict__ CBb,
                              float* __restrict__ C2, int dims, int totalCodes)
{
  int code = blockIdx.x * blockDim.x + threadIdx.x;
  if (code >= totalCodes) return;
  const float* row = CB + (size_t)code * dims;
  u16* out = CBb + (size_t)code * dims;
  float s = 0.f;
  for (int d = 0; d < dims; ++d) {
    float v = row[d];
    s += v * v;
    out[d] = f32_to_bf16(v);
  }
  C2[code] = s;
}

// ---------------------------------------------------------------------------
// RVQ argmin: block = 16 tokens x 8 waves.  The 16x512 bf16 residual tile
// (contiguous 16KB, token-major) is DMA'd into LDS once per block via the
// Tensor Data Mover (tensor_load_to_lds + s_wait_tensorcnt), with the D#
// pad feature inserting 16B per 1KB row so the LDS row stride is 1040B
// (= 260 dwords -> 16 B-fragment lanes land on 16 distinct banks).
// Each wave then sweeps 128 codes with WMMA reading B from LDS;
// score = |c|^2 - 2<r,c> (token norm dropped: argmin-invariant);
// shfl+LDS min-reduce; write int idx and float idx into d_out tail.
// ---------------------------------------------------------------------------
#define RVQ_ROW 520   // 512 dims + 8 u16 pad (1040B stride)

__global__ void rvq_argmin_kernel(const u16* __restrict__ R, const u16* __restrict__ CB,
                                  const float* __restrict__ C2, int* __restrict__ idxOut,
                                  float* __restrict__ idxF, int q, int nCodes, int dims)
{
  __shared__ u16 tile[16 * RVQ_ROW];
  __shared__ float sScore[8][16];
  __shared__ int   sCode[8][16];
  int lane = threadIdx.x & 31;
  int w    = threadIdx.x >> 5;
  int hi   = (lane >> 4) & 1;
  int lcol = lane & 15;
  int tok_base = blockIdx.x * 16;
  const u16* Rblk = R + (size_t)tok_base * dims;

#if HAVE_TDM
  if (threadIdx.x < 32) {                // wave 0 issues one TDM load
    uint64_t ga = (uint64_t)(uintptr_t)Rblk;
    unsigned lds_off = (unsigned)(uintptr_t)(&tile[0]);
    u32x4 g0;
    g0[0] = 1u;                                         // count=1, user D#
    g0[1] = lds_off;                                    // lds_addr
    g0[2] = (unsigned)(ga & 0xffffffffu);               // global_addr[31:0]
    g0[3] = (unsigned)((ga >> 32) & 0x1ffffffu)         // global_addr[56:32]
          | 0x80000000u;                                // type=2 (image)
    i32x8 g1;
    g1[0] = (3 << 16)                                   // data_size = 8B
          | (1 << 20)                                   // pad_enable
          | (7 << 22)                                   // pad_interval = 256 dw (1KB)
          | (3 << 25);                                  // pad_amount = 4 dw (16B)
    g1[1] = (int)(2048u << 16);                         // tensor_dim0 = 2048 (8B elems)
    g1[2] = (1 << 16);                                  // tensor_dim1 = 1
    g1[3] = (int)(2048u << 16);                         // tile_dim0 = 2048
    g1[4] = 0;                                          // tile_dim1/2 = 0 (1D)
    g1[5] = 2048;                                       // tensor_dim0_stride
    g1[6] = 0;
    g1[7] = 0;
    i32x4 gz = {0, 0, 0, 0};
#if defined(__clang_major__) && (__clang_major__ >= 23)
    i32x8 gz8 = {0, 0, 0, 0, 0, 0, 0, 0};
    __builtin_amdgcn_tensor_load_to_lds(g0, g1, gz, gz, gz8, 0);
#else
    __builtin_amdgcn_tensor_load_to_lds(g0, g1, gz, gz, 0);
#endif
    __builtin_amdgcn_s_wait_tensorcnt(0);
  }
  __syncthreads();
#else
  {
    int rowv4 = dims >> 3;                 // uint4 chunks per row
    for (int i = threadIdx.x; i < 16 * rowv4; i += blockDim.x) {
      int row  = i / rowv4;
      int colv = i % rowv4;
      ((uint4*)(tile + row * RVQ_ROW))[colv] = ((const uint4*)(Rblk + (size_t)row * dims))[colv];
    }
  }
  __syncthreads();
#endif

  const u16* rrow = tile + lcol * RVQ_ROW;

  float best = 3.4e38f;
  int bestc = 0;
  int passes = nCodes >> 7;      // 128 codes per pass across 8 waves
  for (int pass = 0; pass < passes; ++pass) {
    int code_base = pass * 128 + w * 16;
    v8f acc = (v8f){0.f,0.f,0.f,0.f,0.f,0.f,0.f,0.f};
    const u16* crow = CB + (size_t)(code_base + lcol) * dims;
    for (int kb = 0; kb < dims; kb += 32) {
      V16 a;
      const u16* ab = crow + kb + hi * 8;
      a.u[0] = *(const uint4*)(ab);
      a.u[1] = *(const uint4*)(ab + 16);
      V16 b;
      const u16* bb = rrow + kb + hi * 8;
      b.u[0] = *(const uint4*)(bb);
      b.u[1] = *(const uint4*)(bb + 16);
      acc = __builtin_amdgcn_wmma_f32_16x16x32_bf16(
          false, a.v, false, b.v, (short)0, acc, false, false);
    }
#pragma unroll
    for (int r = 0; r < 8; ++r) {
      int code = code_base + hi * 8 + r;
      float s = C2[code] - 2.0f * acc[r];
      if (s < best || (s == best && code < bestc)) { best = s; bestc = code; }
    }
  }
  float ob = __shfl_xor(best, 16, 32);
  int   oc = __shfl_xor(bestc, 16, 32);
  if (ob < best || (ob == best && oc < bestc)) { best = ob; bestc = oc; }
  if (lane < 16) { sScore[w][lcol] = best; sCode[w][lcol] = bestc; }
  __syncthreads();
  if (threadIdx.x < 16) {
    float bb = sScore[0][threadIdx.x];
    int   bc = sCode[0][threadIdx.x];
    for (int ww = 1; ww < 8; ++ww) {
      float s = sScore[ww][threadIdx.x];
      int   c = sCode[ww][threadIdx.x];
      if (s < bb || (s == bb && c < bc)) { bb = s; bc = c; }
    }
    int tok = tok_base + (int)threadIdx.x;
    idxOut[tok] = bc;
    idxF[(size_t)tok * 4 + q] = (float)bc;   // indices shape (N, L, 4)
  }
}

// ---------------------------------------------------------------------------
// RVQ residual/qout update: quant = cb[idx]; qout += quant; resid -= quant.
// ---------------------------------------------------------------------------
__global__ void rvq_update_kernel(const int* __restrict__ idx, const float* __restrict__ CBF,
                                  float* __restrict__ residf, u16* __restrict__ residb,
                                  float* __restrict__ qout, int dims)
{
  int tok = blockIdx.x;
  int c = idx[tok];
  const float* crow = CBF + (size_t)c * dims;
  size_t base = (size_t)tok * dims;
  for (int d = threadIdx.x; d < dims; d += blockDim.x) {
    float quant = crow[d];
    float r = residf[base + d] - quant;
    residf[base + d] = r;
    residb[base + d] = f32_to_bf16(r);
    qout[base + d] += quant;
  }
}

__global__ void zero_kernel(float* __restrict__ p, size_t n)
{
  size_t i = (size_t)blockIdx.x * blockDim.x + threadIdx.x;
  if (i < n) p[i] = 0.f;
}

__global__ void cast_bf16_kernel(const float* __restrict__ in, u16* __restrict__ out, size_t n)
{
  size_t i = (size_t)blockIdx.x * blockDim.x + threadIdx.x;
  if (i < n) out[i] = f32_to_bf16(in[i]);
}

// ---------------------------------------------------------------------------
// Final 1x1 conv 64->1 plus crop/pad from L=12287 to 12288.
// ---------------------------------------------------------------------------
__global__ void final_kernel(const u16* __restrict__ R, const float* __restrict__ W,
                             const float* __restrict__ B, float* __restrict__ out,
                             int Lin, int Lout, int C, int N)
{
  size_t i = (size_t)blockIdx.x * blockDim.x + threadIdx.x;
  if (i >= (size_t)N * Lout) return;
  int lo = (int)(i % Lout);
  int n  = (int)(i / Lout);
  float acc = 0.f;
  if (lo < Lin) {
    acc = B[0];
    const u16* row = R + ((size_t)n * Lin + lo) * C;
    for (int ci = 0; ci < C; ++ci) acc += W[ci] * bf16_to_f32(row[ci]);
  }
  out[i] = acc;
}

// ---------------------------------------------------------------------------
// Host driver
// ---------------------------------------------------------------------------
extern "C" void kernel_launch(void* const* d_in, const int* in_sizes, int n_in,
                              void* d_out, int out_size, void* d_ws, size_t ws_size,
                              hipStream_t stream)
{
  (void)in_sizes; (void)n_in; (void)out_size; (void)ws_size;
  char* ws = (char*)d_ws;
  // fixed 256-aligned workspace layout (~376 MB)
  u16*   bufA   = (u16*)  (ws + 0ull);
  u16*   bufB   = (u16*)  (ws + 50331648ull);
  float* yf     = (float*)(ws + 100663296ull);
  float* residf = (float*)(ws + 201326592ull);
  u16*   residb = (u16*)  (ws + 268435456ull);
  float* qout   = (float*)(ws + 301989888ull);
  u16*   cbb    = (u16*)  (ws + 369098752ull);
  float* c2     = (float*)(ws + 373293056ull);
  int*   idxb   = (int*)  (ws + 373309440ull);
  float* stats  = (float*)(ws + 373440512ull);
  u16*   wb     = (u16*)  (ws + 373444608ull);

  const float* x = (const float*)d_in[0];
  float* recon = (float*)d_out;
  float* idxF  = recon + 32 * 12288;          // indices tail of d_out

  // ---- encoder block 1 (Cin=1 direct) ----
  {
    size_t tot = (size_t)32 * 12288 * 64;
    enc1_kernel<<<(unsigned)((tot + 255) / 256), 256, 0, stream>>>(
        x, (const float*)d_in[1], (const float*)d_in[2], yf, 12288, 32, 64, 5, 2);
    bn_stats_kernel<<<64, 256, 0, stream>>>(yf, stats, 64, 32 * 12288);
    bn_apply_kernel<<<(unsigned)((tot + 255) / 256), 256, 0, stream>>>(
        yf, stats, (const float*)d_in[3], (const float*)d_in[4],
        bufA, nullptr, 64, tot);
  }

  struct Layer {
    int wi, Cin, Cout, K, stride, pad, Lin, Lout, isT;
    u16* Xb; u16* Ob; float* Of;
  };
  Layer Ls[7] = {
    { 5,  64,  64,  5, 1, 2, 12288, 12288, 0, bufA, bufB,   nullptr},
    { 9,  64, 128,  9, 3, 4, 12288,  4096, 0, bufB, bufA,   nullptr},
    {13, 128, 128,  9, 2, 4,  4096,  2048, 0, bufA, bufB,   nullptr},
    {17, 128, 512,  5, 2, 2,  2048,  1024, 0, bufB, residb, residf},
    {21, 512, 256,  8, 2, 3,  1024,  2048, 1, bufA, bufB,   nullptr},
    {25, 256, 128, 12, 2, 5,  2048,  4096, 1, bufB, bufA,   nullptr},
    {29, 128,  64, 18, 3, 8,  4096, 12287, 1, bufA, bufB,   nullptr},
  };

  auto runLayer = [&](const Layer& L) {
    size_t wtot = (size_t)L.K * L.Cout * L.Cin;
    wprep_kernel<<<(unsigned)((wtot + 255) / 256), 256, 0, stream>>>(
        (const float*)d_in[L.wi], wb, L.Cout, L.Cin, L.K, L.isT, wtot);
    int totTok = 32 * L.Lout;
    int waves  = (L.Cout / 16) * ((totTok + 63) / 64);
    conv_wmma_kernel<<<(unsigned)((waves + 7) / 8), 256, 0, stream>>>(
        L.Xb, wb, (const float*)d_in[L.wi + 1], yf,
        L.Cin, L.Cout, L.Lin, L.Lout, totTok, L.K, L.stride, L.pad, L.isT);
    bn_stats_kernel<<<L.Cout, 256, 0, stream>>>(yf, stats, L.Cout, totTok);
    size_t tot = (size_t)totTok * L.Cout;
    bn_apply_kernel<<<(unsigned)((tot + 255) / 256), 256, 0, stream>>>(
        yf, stats, (const float*)d_in[L.wi + 2], (const float*)d_in[L.wi + 3],
        L.Ob, L.Of, L.Cout, tot);
  };

  // encoder blocks 2..5 (WMMA)
  for (int i = 0; i < 4; ++i) runLayer(Ls[i]);

  // ---- RVQ ----
  const float* cbf = (const float*)d_in[35];
  zero_kernel<<<65536, 256, 0, stream>>>(qout, (size_t)32768 * 512);
  cbprep_kernel<<<16, 256, 0, stream>>>(cbf, cbb, c2, 512, 4096);
  for (int q = 0; q < 4; ++q) {
    rvq_argmin_kernel<<<2048, 256, 0, stream>>>(
        residb, cbb + (size_t)q * 1024 * 512, c2 + q * 1024,
        idxb, idxF, q, 1024, 512);
    rvq_update_kernel<<<32768, 256, 0, stream>>>(
        idxb, cbf + (size_t)q * 1024 * 512, residf, residb, qout, 512);
  }
  cast_bf16_kernel<<<65536, 256, 0, stream>>>(qout, bufA, (size_t)32768 * 512);

  // decoder blocks (WMMA, transposed conv)
  for (int i = 4; i < 7; ++i) runLayer(Ls[i]);

  // final 1x1 conv + pad to 12288
  {
    size_t tot = (size_t)32 * 12288;
    final_kernel<<<(unsigned)((tot + 255) / 256), 256, 0, stream>>>(
        bufB, (const float*)d_in[33], (const float*)d_in[34], recon,
        12287, 12288, 64, 32);
  }
}
